// DiscoveryNet_15504831939325
// MI455X (gfx1250) — matrix operand order repaired and checked by reading, not compiled, not run on hardware
//
#include <hip/hip_runtime.h>
#include <hip/hip_bf16.h>

typedef __attribute__((ext_vector_type(16))) __bf16 v16bf;
typedef __attribute__((ext_vector_type(8)))  float  v8f;

#define SWZ_X16 0x401f
#define SWZ_X8  0x201f
#define SWZ_X4  0x101f
#define SWZ_X2  0x081f
#define SWZ_X1  0x041f

#define NN 1024
#define HH 64

__device__ __forceinline__ unsigned pk_bf16(float a, float b) {
  __bf16 ha = (__bf16)a, hb = (__bf16)b;
  unsigned short ua = __builtin_bit_cast(unsigned short, ha);
  unsigned short ub = __builtin_bit_cast(unsigned short, hb);
  return (unsigned)ua | ((unsigned)ub << 16);
}

#define DSWZ_F(x, imm) \
  __builtin_bit_cast(float, __builtin_amdgcn_ds_swizzle(__builtin_bit_cast(int, (x)), (imm)))

union BV16 { v16bf v; unsigned u[8]; };

// SiLU with hardware reciprocal: denominator is in (1, inf), so v_rcp_f32's
// ~1 ulp error is negligible vs the bf16 quantization already in the pipeline.
__device__ __forceinline__ float silu(float x) {
  return x * __builtin_amdgcn_rcpf(1.0f + __expf(-x));
}

// Convert per-lane C-fragments (4 tiles of Hᵀ, already SiLU'd, f32) into the two
// K=32 bf16 B-fragments of the next layer. C layout: lane holds column(pair)=lane%16,
// rows m = r + 8*(lane/16). B layout: lane<16 holds K=0..15 of its column,
// lane>=16 holds K=16..31. Row-halves are exchanged with ds_swizzle SWAPX16.
__device__ __forceinline__ void makeB(const float s[4][8], int hi, v16bf& B0, v16bf& B1) {
  unsigned own[4][4], swz[4][4];
#pragma unroll
  for (int t = 0; t < 4; ++t)
#pragma unroll
    for (int q = 0; q < 4; ++q)
      own[t][q] = pk_bf16(s[t][2 * q], s[t][2 * q + 1]);
#pragma unroll
  for (int t = 0; t < 4; ++t)
#pragma unroll
    for (int q = 0; q < 4; ++q)
      swz[t][q] = (unsigned)__builtin_amdgcn_ds_swizzle((int)own[t][q], SWZ_X16);
  BV16 u0, u1;
#pragma unroll
  for (int q = 0; q < 4; ++q) {
    u0.u[q]     = hi ? swz[1][q] : own[0][q];
    u0.u[4 + q] = hi ? own[1][q] : swz[0][q];
    u1.u[q]     = hi ? swz[3][q] : own[2][q];
    u1.u[4 + q] = hi ? own[3][q] : swz[2][q];
  }
  B0 = u0.v; B1 = u1.v;
}

__global__ void __launch_bounds__(256)
discovery_force_kernel(const float* __restrict__ p,
                       const float* __restrict__ pscale,
                       const float* __restrict__ W1, const float* __restrict__ b1,
                       const float* __restrict__ W2, const float* __restrict__ b2,
                       const float* __restrict__ W3, const float* __restrict__ b3,
                       const float* __restrict__ mu, const float* __restrict__ sd,
                       float* __restrict__ out) {
  __shared__ float sp[NN * 3];

  const int tid  = threadIdx.x;
  const int wid  = blockIdx.x * 8 + (tid >> 5);  // global wave id = b*N + i
  const int lane = tid & 31;
  const int lo   = lane & 15;
  const int hi   = lane >> 4;
  const int b    = wid >> 10;
  const int i    = wid & (NN - 1);

  // Stage this batch's positions into LDS (all 8 waves in a block share b).
  for (int k = tid; k < NN * 3; k += 256) sp[k] = p[b * (NN * 3) + k];
  __syncthreads();

  const float ps = pscale[0];

  // ---- Build weight fragments (A-operands, M = output feature, transposed MLP) ----
  float rstd[6], muv[6];
#pragma unroll
  for (int k = 0; k < 6; ++k) { rstd[k] = 1.0f / sd[k]; muv[k] = mu[k]; }

  // W1ᵀ fragments: K=0..5 = normalized weights, K=6 carries the effective bias
  // (paired with a constant-1 feature). Lanes>=16 (K=16..31) are zero.
  v16bf w1f[4];
#pragma unroll
  for (int t = 0; t < 4; ++t) {
    v16bf a = {};
    if (hi == 0) {
      int f = 16 * t + lo;
      float bias = b1[f];
#pragma unroll
      for (int k = 0; k < 6; ++k) {
        float w = W1[k * HH + f] * rstd[k];
        a[k] = (__bf16)w;
        bias -= muv[k] * w;
      }
      a[6] = (__bf16)bias;  // bias row, multiplied by constant-1 feature
    }
    w1f[t] = a;
  }

  // W2ᵀ fragments: 4 M-tiles x 2 K-chunks.
  v16bf w2f[4][2];
#pragma unroll
  for (int t = 0; t < 4; ++t)
#pragma unroll
    for (int c = 0; c < 2; ++c) {
      v16bf a;
      int f  = 16 * t + lo;
      int kb = 32 * c + 16 * hi;
#pragma unroll
      for (int j = 0; j < 8; ++j) {
        a[j]     = (__bf16)W2[(kb + j) * HH + f];
        a[8 + j] = (__bf16)W2[(kb + 8 + j) * HH + f];
      }
      w2f[t][c] = a;
    }

  // W3ᵀ fragments: only M-row 0 is non-zero (lanes 0 and 16).
  v16bf w3f[2];
#pragma unroll
  for (int c = 0; c < 2; ++c) {
    v16bf a = {};
    if (lo == 0) {
      int kb = 32 * c + 16 * hi;
#pragma unroll
      for (int j = 0; j < 8; ++j) {
        a[j]     = (__bf16)W3[kb + j];
        a[8 + j] = (__bf16)W3[kb + 8 + j];
      }
    }
    w3f[c] = a;
  }

  float b2v[4][8];
#pragma unroll
  for (int t = 0; t < 4; ++t)
#pragma unroll
    for (int r = 0; r < 8; ++r) b2v[t][r] = b2[16 * t + r + 8 * hi];
  const float b3v = b3[0];

  const float pix = sp[3 * i + 0], piy = sp[3 * i + 1], piz = sp[3 * i + 2];

  float ax = 0.0f, ay = 0.0f, az = 0.0f;

  for (int j0 = 0; j0 < NN; j0 += 16) {
    const int j = j0 + lo;
    float dx = pix - sp[3 * j + 0];
    float dy = piy - sp[3 * j + 1];
    float dz = piz - sp[3 * j + 2];
    float sq = dx * dx + dy * dy + dz * dz;
    float d  = sqrtf(sq);
    float dp = d * ps;
    float dc = fminf(fmaxf(dp, 0.4f), 10.0f);
    float inv   = __builtin_amdgcn_rcpf(dc);   // dc in [0.4, 10]: well conditioned
    float inv2  = inv * inv;
    float inv3  = inv2 * inv;
    float inv6  = inv3 * inv3;
    float inv12 = inv6 * inv6;

    // Feature B-fragment (featᵀ): K = feature, N = pair. Only lanes<16 (K=0..15) used.
    v16bf fB = {};
    if (hi == 0) {
      fB[0] = (__bf16)dc;           fB[1] = (__bf16)inv;
      fB[2] = (__bf16)inv6;         fB[3] = (__bf16)inv12;
      fB[4] = (__bf16)(inv6 * inv); fB[5] = (__bf16)(inv12 * inv);
      fB[6] = (__bf16)1.0f;         // multiplies folded bias row of W1ᵀ
    }

    // Layer 1: H1ᵀ = silu(W1ᵀ · featᵀ)
    float s1[4][8];
#pragma unroll
    for (int t = 0; t < 4; ++t) {
      v8f c = {};
      c = __builtin_amdgcn_wmma_f32_16x16x32_bf16(false, w1f[t], false, fB,
                                                  (short)0, c, false, false);
#pragma unroll
      for (int r = 0; r < 8; ++r) s1[t][r] = silu(c[r]);
    }
    v16bf B0, B1;
    makeB(s1, hi, B0, B1);

    // Layer 2: H2ᵀ = silu(W2ᵀ · H1ᵀ + b2)
    float s2[4][8];
#pragma unroll
    for (int t = 0; t < 4; ++t) {
      v8f c;
#pragma unroll
      for (int r = 0; r < 8; ++r) c[r] = b2v[t][r];
      c = __builtin_amdgcn_wmma_f32_16x16x32_bf16(false, w2f[t][0], false, B0,
                                                  (short)0, c, false, false);
      c = __builtin_amdgcn_wmma_f32_16x16x32_bf16(false, w2f[t][1], false, B1,
                                                  (short)0, c, false, false);
#pragma unroll
      for (int r = 0; r < 8; ++r) s2[t][r] = silu(c[r]);
    }
    v16bf C0, C1;
    makeB(s2, hi, C0, C1);

    // Layer 3: magᵀ = W3ᵀ · H2ᵀ + b3   (result row 0 -> lane = pair, VGPR 0)
    v8f c3;
#pragma unroll
    for (int r = 0; r < 8; ++r) c3[r] = b3v;
    c3 = __builtin_amdgcn_wmma_f32_16x16x32_bf16(false, w3f[0], false, C0,
                                                 (short)0, c3, false, false);
    c3 = __builtin_amdgcn_wmma_f32_16x16x32_bf16(false, w3f[1], false, C1,
                                                 (short)0, c3, false, false);
    float mag = c3[0];

    float denom = fmaxf(d, 0.001f);  // denom >= 1e-3, rcp is safe
    float w = ((hi == 0) && (j != i)) ? (mag * __builtin_amdgcn_rcpf(denom)) : 0.0f;
    ax += w * dx;
    ay += w * dy;
    az += w * dz;
  }

  // Cross-lane reduction of the 16 per-pair partials (lanes>=16 contribute 0).
  ax += DSWZ_F(ax, SWZ_X16); ay += DSWZ_F(ay, SWZ_X16); az += DSWZ_F(az, SWZ_X16);
  ax += DSWZ_F(ax, SWZ_X8);  ay += DSWZ_F(ay, SWZ_X8);  az += DSWZ_F(az, SWZ_X8);
  ax += DSWZ_F(ax, SWZ_X4);  ay += DSWZ_F(ay, SWZ_X4);  az += DSWZ_F(az, SWZ_X4);
  ax += DSWZ_F(ax, SWZ_X2);  ay += DSWZ_F(ay, SWZ_X2);  az += DSWZ_F(az, SWZ_X2);
  ax += DSWZ_F(ax, SWZ_X1);  ay += DSWZ_F(ay, SWZ_X1);  az += DSWZ_F(az, SWZ_X1);

  if (lane == 0) {
    out[wid * 3 + 0] = ax;
    out[wid * 3 + 1] = ay;
    out[wid * 3 + 2] = az;
  }
}

extern "C" void kernel_launch(void* const* d_in, const int* in_sizes, int n_in,
                              void* d_out, int out_size, void* d_ws, size_t ws_size,
                              hipStream_t stream) {
  (void)n_in; (void)out_size; (void)d_ws; (void)ws_size;
  const float* p  = (const float*)d_in[0];
  const float* ps = (const float*)d_in[1];
  const float* W1 = (const float*)d_in[2];
  const float* b1 = (const float*)d_in[3];
  const float* W2 = (const float*)d_in[4];
  const float* b2 = (const float*)d_in[5];
  const float* W3 = (const float*)d_in[6];
  const float* b3 = (const float*)d_in[7];
  const float* mu = (const float*)d_in[8];
  const float* sd = (const float*)d_in[9];
  float* out = (float*)d_out;

  const int BN = in_sizes[0] / 3;   // B*N waves, one wave per output row
  const int nblocks = BN / 8;       // 8 waves (256 threads) per block
  discovery_force_kernel<<<nblocks, 256, 0, stream>>>(p, ps, W1, b1, W2, b2,
                                                      W3, b3, mu, sd, out);
}
